// MultiheadAttention_81320910782906
// MI455X (gfx1250) — matrix operand order repaired and checked by reading, not compiled, not run on hardware
//
#include <hip/hip_runtime.h>

// ---------------------------------------------------------------------------
// MHA for MI455X (gfx1250): bf16 WMMA everywhere, fp32 accumulate.
// B=2, S=2048, H=1024, NH=16, HD=64.
// GEMMs: 128x128 block tile, double-buffered LDS filled with
// global_load_async_to_lds_b128 (ASYNCcnt), 8 WMMAs/wave/K-step.
// Attention: flash-style, double-buffered V tile, 1 barrier per 32-key step.
// ---------------------------------------------------------------------------

#define B_   2
#define S_   2048
#define H_   1024
#define NH_  16
#define HD_  64
#define M_   (B_ * S_)        // 4096 rows
#define N3_  (3 * H_)         // 3072
#define NEGV (-1.0e10f)

typedef __attribute__((ext_vector_type(16))) __bf16 bf16x16;
typedef __attribute__((ext_vector_type(8)))  __bf16 bf16x8;
typedef __attribute__((ext_vector_type(8)))  float  f32x8;

static __device__ __forceinline__ __bf16 to_bf16(float f) { return (__bf16)f; }

#define WMMA_BF16(A, Bv, C) \
  __builtin_amdgcn_wmma_f32_16x16x32_bf16(false, (A), false, (Bv), (short)0, (C), false, false)

// Async global->LDS 16B copy (per lane). VDST carries the LDS byte offset
// (flat-pointer low 32 bits, per ISA LDS aperture rule), VADDR the 64-bit
// global address. Tracked by ASYNCcnt.
static __device__ __forceinline__ void async_b128(void* lds, const void* g) {
  asm volatile("global_load_async_to_lds_b128 %0, %1, off"
               :: "v"((unsigned int)(unsigned long long)lds), "v"(g)
               : "memory");
}
static __device__ __forceinline__ void wait_async0() {
  asm volatile("s_wait_asynccnt 0" ::: "memory");
}

// A-matrix fragment (16x32 bf16, M x K), per ISA layout:
// lane<16: elems 0..7 = K[0..7], 8..15 = K[16..23] of row M=lane
// lane>=16: elems 0..7 = K[8..15], 8..15 = K[24..31] of row M=lane-16
static __device__ __forceinline__ bf16x16 load_frag_a(const __bf16* base, int lda, int lane) {
  const int m    = lane & 15;
  const int koff = (lane & 16) ? 8 : 0;
  const __bf16* p = base + (size_t)m * lda;
  bf16x8 lo = *(const bf16x8*)(p + koff);
  bf16x8 hi = *(const bf16x8*)(p + 16 + koff);
  return __builtin_shufflevector(lo, hi, 0,1,2,3,4,5,6,7,8,9,10,11,12,13,14,15);
}

// B-matrix fragment (32x16 bf16, K x N), operand stored K-contiguous per column
// (baseT[n*ldb + k] == B[k][n]); lanes 0..15: K=0..7,16..23, lanes 16..31: K=8..15,24..31.
static __device__ __forceinline__ bf16x16 load_frag_b(const __bf16* baseT, int ldb, int lane) {
  const int n    = lane & 15;
  const int koff = (lane & 16) ? 8 : 0;
  const __bf16* p = baseT + (size_t)n * ldb + koff;
  bf16x8 lo = *(const bf16x8*)(p);
  bf16x8 hi = *(const bf16x8*)(p + 16);
  return __builtin_shufflevector(lo, hi, 0,1,2,3,4,5,6,7,8,9,10,11,12,13,14,15);
}

// ---------------------------------------------------------------------------
// Conversion kernels
// ---------------------------------------------------------------------------
__global__ void convert_bf16_kernel(const float* __restrict__ src,
                                    __bf16* __restrict__ dst, int n) {
  int i = blockIdx.x * blockDim.x + threadIdx.x;
  if (i < n) dst[i] = to_bf16(src[i]);
}

// dst[c*R + r] = src[r*C + c]  (N-major weights -> K-contiguous B-fragments)
__global__ void transpose_bf16_kernel(const float* __restrict__ src,
                                      __bf16* __restrict__ dst, int R, int C) {
  int i = blockIdx.x * blockDim.x + threadIdx.x;
  if (i < R * C) {
    int r = i / C, c = i % C;
    dst[(size_t)c * R + r] = to_bf16(src[(size_t)r * C + c]);
  }
}

// ---------------------------------------------------------------------------
// GEMM core: C[128x128] block tile, K=1024, 256 threads = 8 waves (4x2).
// Wave w (wm=w&3, wn=w>>2) owns rows [wm*32,+32) x cols [wn*64,+64):
// 8 accumulators, 8 WMMAs per 32-K step from 6 fragment loads.
// Double-buffered LDS, async fill, one barrier per step.
// A: [M,1024] row-major bf16. Bt: [N,1024] (N-major) bf16.
// ---------------------------------------------------------------------------
#define ATILE (128 * 32)
#define BTILE (128 * 32)

__device__ __forceinline__ void stage_tiles(const __bf16* __restrict__ A,
                                            const __bf16* __restrict__ Bt,
                                            int m0, int n0, int k0,
                                            __bf16* Adst, __bf16* Bdst, int t) {
  const int row = t >> 1, col = (t & 1) * 16;     // 128 rows x 32 cols, 2x16B/thread
  const __bf16* ga = A + (size_t)(m0 + row) * H_ + k0 + col;
  async_b128(Adst + row * 32 + col,     ga);
  async_b128(Adst + row * 32 + col + 8, ga + 8);
  const __bf16* gb = Bt + (size_t)(n0 + row) * H_ + k0 + col;
  async_b128(Bdst + row * 32 + col,     gb);
  async_b128(Bdst + row * 32 + col + 8, gb + 8);
}

__device__ __forceinline__ void gemm_core_128x128(
    const __bf16* __restrict__ A, const __bf16* __restrict__ Bt,
    int m0, int n0, __bf16* Asub, __bf16* Bsub, f32x8 acc[8]) {
  const int t    = (int)threadIdx.x;
  const int lane = t & 31;
  const int wave = t >> 5;
  const int wm   = wave & 3;       // row group (32 rows)
  const int wn   = wave >> 2;      // col group (64 cols)

  // prologue: fill buffer 0
  stage_tiles(A, Bt, m0, n0, 0, Asub, Bsub, t);
  wait_async0();
  __syncthreads();

  for (int k0 = 0, it = 0; k0 < H_; k0 += 32, ++it) {
    __bf16* Acur = Asub + (it & 1) * ATILE;
    __bf16* Bcur = Bsub + (it & 1) * BTILE;
    if (k0 + 32 < H_) {  // overlap next fill with current WMMAs
      stage_tiles(A, Bt, m0, n0, k0 + 32,
                  Asub + ((it + 1) & 1) * ATILE,
                  Bsub + ((it + 1) & 1) * BTILE, t);
    }
    bf16x16 a0 = load_frag_a(Acur + (wm * 32 + 0)  * 32, 32, lane);
    bf16x16 a1 = load_frag_a(Acur + (wm * 32 + 16) * 32, 32, lane);
#pragma unroll
    for (int ni = 0; ni < 4; ++ni) {
      bf16x16 b = load_frag_b(Bcur + (wn * 64 + ni * 16) * 32, 32, lane);
      acc[ni]     = WMMA_BF16(a0, b, acc[ni]);
      acc[4 + ni] = WMMA_BF16(a1, b, acc[4 + ni]);
    }
    wait_async0();
    __syncthreads();
  }
}

// ---------------------------------------------------------------------------
// QKV projection: [4096,1024] x [1024,3072] + b_qkv, scatter into
// qkv[3][B*NH][S][HD] bf16 (head-major for the attention kernel).
// ---------------------------------------------------------------------------
__global__ __launch_bounds__(256)
void gemm_qkv_kernel(const __bf16* __restrict__ A, const __bf16* __restrict__ Bt,
                     const float* __restrict__ bias, __bf16* __restrict__ qkv) {
  __shared__ __attribute__((aligned(16))) __bf16 Asub[2 * ATILE];
  __shared__ __attribute__((aligned(16))) __bf16 Bsub[2 * BTILE];
  const int lane = (int)threadIdx.x & 31;
  const int wave = (int)threadIdx.x >> 5;
  const int n0 = blockIdx.x * 128;
  const int m0 = blockIdx.y * 128;
  f32x8 acc[8] = {{}, {}, {}, {}, {}, {}, {}, {}};
  gemm_core_128x128(A, Bt, m0, n0, Asub, Bsub, acc);

  const int nlane = lane & 15;
  const int hi8   = (lane >> 4) * 8;
  const int wm = wave & 3, wn = wave >> 2;
#pragma unroll
  for (int mi = 0; mi < 2; ++mi) {
#pragma unroll
    for (int ni = 0; ni < 4; ++ni) {
      const f32x8& a = acc[mi * 4 + ni];
      const int gn    = n0 + wn * 64 + ni * 16 + nlane;
      const float bv  = bias[gn];
      const int which = gn >> 10;          // 0=Q 1=K 2=V
      const int rem   = gn & 1023;
      const int hh    = rem >> 6;          // head
      const int d     = rem & 63;          // dim in head
#pragma unroll
      for (int r = 0; r < 8; ++r) {
        const int gm = m0 + wm * 32 + mi * 16 + r + hi8;
        const int bb = gm >> 11;           // batch
        const int ss = gm & 2047;          // seq pos
        qkv[(size_t)which * (B_ * NH_ * S_ * HD_) +
            ((size_t)(bb * NH_ + hh) * S_ + ss) * HD_ + d] = to_bf16(a[r] + bv);
      }
    }
  }
}

// ---------------------------------------------------------------------------
// Flash attention. Block = 128 threads (4 waves), each wave owns 16 q-rows.
// grid.x = (B*NH) * (S/64).  Per 32-key step: 4 score WMMAs + 4 PV WMMAs,
// double-buffered V tile, single workgroup barrier per step; P transpose is
// intra-wave via LDS (same-wave DS ops are in-order) + wavefront fence.
// ---------------------------------------------------------------------------
#define VTILE (HD_ * 32)

__device__ __forceinline__ void stage_v(const __bf16* __restrict__ Vh, int k0,
                                        __bf16* dst, int t) {
  const int key = t >> 2;            // 0..31
  const int c0  = (t & 3) * 16;      // 0..48
  const __bf16* vsrc = Vh + (size_t)(k0 + key) * HD_ + c0;
#pragma unroll
  for (int j = 0; j < 16; ++j) dst[(c0 + j) * 32 + key] = vsrc[j];
}

__global__ __launch_bounds__(128)
void attn_kernel(const __bf16* __restrict__ Q, const __bf16* __restrict__ K,
                 const __bf16* __restrict__ V, const int* __restrict__ mask,
                 __bf16* __restrict__ ctx) {
  __shared__ __attribute__((aligned(16))) __bf16 Vt[2 * VTILE];     // [buf][d][key]
  __shared__ __attribute__((aligned(16))) __bf16 Pl[4][16 * 32];    // per-wave P

  const int t    = (int)threadIdx.x;
  const int lane = t & 31;
  const int wave = t >> 5;
  const int bh   = blockIdx.x >> 5;          // 0..31  (b*16 + h)
  const int qblk = blockIdx.x & 31;          // 0..31
  const int b    = bh >> 4;
  const int h    = bh & 15;
  const int q0   = qblk * 64 + wave * 16;

  const __bf16* Qh = Q + (size_t)bh * S_ * HD_;
  const __bf16* Kh = K + (size_t)bh * S_ * HD_;
  const __bf16* Vh = V + (size_t)bh * S_ * HD_;
  const int* mrow  = mask + (size_t)b * S_ * S_;
  const float scale = 0.125f;                // HD^-0.5

  // Q tile (16 x 64) as two A fragments, straight from global (row-major HD).
  bf16x16 qf0 = load_frag_a(Qh + (size_t)q0 * HD_ + 0,  HD_, lane);
  bf16x16 qf1 = load_frag_a(Qh + (size_t)q0 * HD_ + 32, HD_, lane);

  f32x8 o0 = {}, o1 = {}, o2 = {}, o3 = {};
  float mrun[8], lrun[8];
#pragma unroll
  for (int r = 0; r < 8; ++r) { mrun[r] = -3.0e38f; lrun[r] = 0.0f; }

  const int nlane = lane & 15;
  const int hi8   = (lane >> 4) * 8;

  // prologue: stage V tile 0
  stage_v(Vh, 0, Vt, t);
  __syncthreads();

  for (int k0 = 0; k0 < S_; k0 += 32) {
    const int it  = k0 >> 5;
    __bf16* Vcur  = Vt + (it & 1) * VTILE;
    if (k0 + 32 < S_) {
      // stage next V tile into the other buffer (overlaps with compute)
      stage_v(Vh, k0 + 32, Vt + ((it + 1) & 1) * VTILE, t);
      // prefetch next K tile (global_prefetch_b8): 32 lanes x 128B = 4KB
      __builtin_prefetch(Kh + (size_t)(k0 + 32) * HD_ + lane * 64, 0, 3);
    }

    // scores: s0 = keys k0..k0+15, s1 = keys k0+16..k0+31 (K^T frags contiguous)
    f32x8 s0 = {}, s1 = {};
    {
      bf16x16 kb;
      kb = load_frag_b(Kh + (size_t)(k0 + 0)  * HD_ + 0,  HD_, lane); s0 = WMMA_BF16(qf0, kb, s0);
      kb = load_frag_b(Kh + (size_t)(k0 + 0)  * HD_ + 32, HD_, lane); s0 = WMMA_BF16(qf1, kb, s0);
      kb = load_frag_b(Kh + (size_t)(k0 + 16) * HD_ + 0,  HD_, lane); s1 = WMMA_BF16(qf0, kb, s1);
      kb = load_frag_b(Kh + (size_t)(k0 + 16) * HD_ + 32, HD_, lane); s1 = WMMA_BF16(qf1, kb, s1);
    }

    // mask + online softmax (rows live across the 16 lanes of each half-wave)
    int mk0[8], mk1[8];
#pragma unroll
    for (int r = 0; r < 8; ++r) {
      const int q = q0 + r + hi8;
      mk0[r] = mrow[(size_t)q * S_ + k0 + nlane];
      mk1[r] = mrow[(size_t)q * S_ + k0 + 16 + nlane];
      s0[r] = mk0[r] ? NEGV : s0[r] * scale;
      s1[r] = mk1[r] ? NEGV : s1[r] * scale;
    }
#pragma unroll
    for (int r = 0; r < 8; ++r) {
      float rm = fmaxf(s0[r], s1[r]);
      rm = fmaxf(rm, __shfl_xor(rm, 1));
      rm = fmaxf(rm, __shfl_xor(rm, 2));
      rm = fmaxf(rm, __shfl_xor(rm, 4));
      rm = fmaxf(rm, __shfl_xor(rm, 8));
      const float mnew = fmaxf(mrun[r], rm);
      const float corr = __expf(mrun[r] - mnew);
      const float p0 = mk0[r] ? 0.0f : __expf(s0[r] - mnew);
      const float p1 = mk1[r] ? 0.0f : __expf(s1[r] - mnew);
      float rs = p0 + p1;
      rs += __shfl_xor(rs, 1);
      rs += __shfl_xor(rs, 2);
      rs += __shfl_xor(rs, 4);
      rs += __shfl_xor(rs, 8);
      lrun[r] = lrun[r] * corr + rs;
      mrun[r] = mnew;
      o0[r] *= corr; o1[r] *= corr; o2[r] *= corr; o3[r] *= corr;
      // stash P (C-layout -> A-layout transpose goes through per-wave LDS)
      __bf16* pl = &Pl[wave][(r + hi8) * 32];
      pl[nlane]      = to_bf16(p0);
      pl[16 + nlane] = to_bf16(p1);
    }
    // same-wave LDS ops execute in order; fence only blocks compiler reordering
    __builtin_amdgcn_fence(__ATOMIC_ACQ_REL, "wavefront");

    bf16x16 pf = load_frag_a(&Pl[wave][0], 32, lane);
    bf16x16 vb;
    vb = load_frag_b(Vcur + 0 * 16 * 32, 32, lane); o0 = WMMA_BF16(pf, vb, o0);
    vb = load_frag_b(Vcur + 1 * 16 * 32, 32, lane); o1 = WMMA_BF16(pf, vb, o1);
    vb = load_frag_b(Vcur + 2 * 16 * 32, 32, lane); o2 = WMMA_BF16(pf, vb, o2);
    vb = load_frag_b(Vcur + 3 * 16 * 32, 32, lane); o3 = WMMA_BF16(pf, vb, o3);

    __syncthreads();   // V buffer swap point (covers next iteration's stage_v)
  }

  // normalize (fully-masked rows: l==0 -> output 0, matching the reference)
#pragma unroll
  for (int r = 0; r < 8; ++r) {
    const int q = q0 + r + hi8;
    const float inv = (lrun[r] > 0.0f) ? (1.0f / lrun[r]) : 0.0f;
    __bf16* dst = ctx + ((size_t)(b * S_ + q)) * H_ + h * HD_;
    dst[0 * 16 + nlane] = to_bf16(o0[r] * inv);
    dst[1 * 16 + nlane] = to_bf16(o1[r] * inv);
    dst[2 * 16 + nlane] = to_bf16(o2[r] * inv);
    dst[3 * 16 + nlane] = to_bf16(o3[r] * inv);
  }
}

// ---------------------------------------------------------------------------
// Output projection: ctx[4096,1024] x W_out[1024,1024] + b_out -> f32 d_out.
// ---------------------------------------------------------------------------
__global__ __launch_bounds__(256)
void gemm_out_kernel(const __bf16* __restrict__ A, const __bf16* __restrict__ Bt,
                     const float* __restrict__ bias, float* __restrict__ out) {
  __shared__ __attribute__((aligned(16))) __bf16 Asub[2 * ATILE];
  __shared__ __attribute__((aligned(16))) __bf16 Bsub[2 * BTILE];
  const int lane = (int)threadIdx.x & 31;
  const int wave = (int)threadIdx.x >> 5;
  const int n0 = blockIdx.x * 128;
  const int m0 = blockIdx.y * 128;
  f32x8 acc[8] = {{}, {}, {}, {}, {}, {}, {}, {}};
  gemm_core_128x128(A, Bt, m0, n0, Asub, Bsub, acc);

  const int nlane = lane & 15;
  const int hi8   = (lane >> 4) * 8;
  const int wm = wave & 3, wn = wave >> 2;
#pragma unroll
  for (int mi = 0; mi < 2; ++mi) {
#pragma unroll
    for (int ni = 0; ni < 4; ++ni) {
      const f32x8& a = acc[mi * 4 + ni];
      const int gn   = n0 + wn * 64 + ni * 16 + nlane;
      const float bv = bias[gn];
#pragma unroll
      for (int r = 0; r < 8; ++r) {
        const int gm = m0 + wm * 32 + mi * 16 + r + hi8;
        out[(size_t)gm * H_ + gn] = a[r] + bv;
      }
    }
  }
}

// ---------------------------------------------------------------------------
// Host launcher
// ---------------------------------------------------------------------------
extern "C" void kernel_launch(void* const* d_in, const int* in_sizes, int n_in,
                              void* d_out, int out_size, void* d_ws, size_t ws_size,
                              hipStream_t stream) {
  const float* inp  = (const float*)d_in[0];   // [B,S,H]
  const int*   mask = (const int*)d_in[1];     // [B,S,S]
  const float* Wqkv = (const float*)d_in[2];   // [H,3H]
  const float* bqkv = (const float*)d_in[3];   // [3H]
  const float* Wout = (const float*)d_in[4];   // [H,H]
  const float* bout = (const float*)d_in[5];   // [H]
  float* out = (float*)d_out;                  // [B,S,H] f32

  // Workspace layout (bf16 buffers), total ~48 MB
  char* ws = (char*)d_ws;
  __bf16* Abf   = (__bf16*)ws;  ws += (size_t)M_ * H_ * 2;        // inp bf16     8 MB
  __bf16* Wqt   = (__bf16*)ws;  ws += (size_t)N3_ * H_ * 2;       // W_qkv^T      6 MB
  __bf16* Wot   = (__bf16*)ws;  ws += (size_t)H_ * H_ * 2;        // W_out^T      2 MB
  __bf16* QKVb  = (__bf16*)ws;  ws += (size_t)3 * B_ * NH_ * S_ * HD_ * 2; //    24 MB
  __bf16* Ctx   = (__bf16*)ws;  ws += (size_t)M_ * H_ * 2;        // attn output  8 MB

  __bf16* Qb = QKVb;
  __bf16* Kb = QKVb + (size_t)B_ * NH_ * S_ * HD_;
  __bf16* Vb = QKVb + (size_t)2 * B_ * NH_ * S_ * HD_;

  // 1) precision conversion / weight transposition
  {
    const int n = M_ * H_;
    convert_bf16_kernel<<<(n + 255) / 256, 256, 0, stream>>>(inp, Abf, n);
  }
  {
    const int n = H_ * N3_;
    transpose_bf16_kernel<<<(n + 255) / 256, 256, 0, stream>>>(Wqkv, Wqt, H_, N3_);
  }
  {
    const int n = H_ * H_;
    transpose_bf16_kernel<<<(n + 255) / 256, 256, 0, stream>>>(Wout, Wot, H_, H_);
  }

  // 2) QKV projection + head-major scatter (grid 24 x 32)
  gemm_qkv_kernel<<<dim3(N3_ / 128, M_ / 128), 256, 0, stream>>>(Abf, Wqt, bqkv, QKVb);

  // 3) flash attention (grid = B*NH * S/64 blocks of 4 waves)
  attn_kernel<<<(B_ * NH_) * (S_ / 64), 128, 0, stream>>>(Qb, Kb, Vb, mask, Ctx);

  // 4) output projection -> f32 (grid 8 x 32)
  gemm_out_kernel<<<dim3(H_ / 128, M_ / 128), 256, 0, stream>>>(Ctx, Wot, bout, out);
}
